// Attention_82068235092457
// MI455X (gfx1250) — matrix operand order repaired and checked by reading, not compiled, not run on hardware
//
#include <hip/hip_runtime.h>
#include <hip/hip_bf16.h>

// ---------------------------------------------------------------------------
// Types for CDNA5 WMMA (wave32): v_wmma_f32_16x16x32_bf16
// ---------------------------------------------------------------------------
typedef __attribute__((ext_vector_type(16))) __bf16 v16bf;
typedef __attribute__((ext_vector_type(8)))  float  v8f;

struct W8  { unsigned int w[8]; };   // 32 bytes -> one v16bf fragment
struct U4x2 { uint4 a, b; };         // 32 bytes -> one v16bf fragment

__device__ __forceinline__ unsigned short f2bf(float f) {
    unsigned int u = __builtin_bit_cast(unsigned int, f);
    unsigned int r = (u + 0x7FFFu + ((u >> 16) & 1u)) >> 16;
    return (unsigned short)r;
}

// ---------------------------------------------------------------------------
// CDNA5 async global->LDS copy (ASYNCcnt-tracked, no VGPR round-trip).
// VDST takes the LDS byte offset (low 32 bits of the generic pointer, which
// for the LDS aperture equals the workgroup-relative LDS address).
// ---------------------------------------------------------------------------
__device__ __forceinline__ void async_copy_b128(void* lds, const void* gaddr) {
    unsigned ldsoff = (unsigned)(unsigned long long)lds;
    asm volatile("global_load_async_to_lds_b128 %0, %1, off"
                 :: "v"(ldsoff), "v"(gaddr)
                 : "memory");
}
__device__ __forceinline__ void wait_async0() {
    asm volatile("s_wait_asynccnt 0x0" ::: "memory");
}

// A-fragment (16x32 bf16, MxK): lane L holds row M=L&15.
// Dword i (i=0..7) holds K pair at offset ((L>>4)&1)*16 + i*2.
__device__ __forceinline__ v16bf load_a_frag(const unsigned short* rowPtr, int lane) {
    const int hi8 = ((lane >> 4) & 1) * 16;
    W8 w;
#pragma unroll
    for (int i = 0; i < 8; ++i) {
        int koff = hi8 + (i * 2);
        w.w[i] = *(const unsigned int*)(rowPtr + koff);
    }
    return __builtin_bit_cast(v16bf, w);
}

// B-fragment (32x16 bf16, KxN) from K-major storage: lane L holds col N=L&15,
// 16 contiguous K values starting at (L>>4)*16.  p points at those 16 halves.
__device__ __forceinline__ v16bf load_b_frag(const unsigned short* p) {
    U4x2 u;
    u.a = *(const uint4*)(p);
    u.b = *(const uint4*)(p + 8);
    return __builtin_bit_cast(v16bf, u);
}

__device__ __forceinline__ v8f wmma_bf16(v16bf a, v16bf b, v8f c) {
    return __builtin_amdgcn_wmma_f32_16x16x32_bf16(false, a, false, b,
                                                   (short)0, c, false, false);
}

// ---------------------------------------------------------------------------
// 0) fp32 -> bf16 conversion (weights)
// ---------------------------------------------------------------------------
__global__ void cvt_bf16_kernel(const float* __restrict__ src,
                                unsigned short* __restrict__ dst, int n) {
    for (int i = blockIdx.x * 256 + threadIdx.x; i < n; i += gridDim.x * 256)
        dst[i] = f2bf(src[i]);
}

// ---------------------------------------------------------------------------
// 1) LayerNorm (per 2048-wide row) -> bf16
// ---------------------------------------------------------------------------
__global__ void ln_kernel(const float* __restrict__ x,
                          const float* __restrict__ gamma,
                          unsigned short* __restrict__ xnb) {
    const int row = blockIdx.x;
    const float* xr = x + (size_t)row * 2048;
    __shared__ float red[256];
    float s = 0.f, s2 = 0.f;
    for (int i = threadIdx.x; i < 2048; i += 256) {
        float v = xr[i];
        s += v; s2 += v * v;
    }
    red[threadIdx.x] = s; __syncthreads();
    for (int o = 128; o; o >>= 1) {
        if (threadIdx.x < o) red[threadIdx.x] += red[threadIdx.x + o];
        __syncthreads();
    }
    float mean = red[0] * (1.0f / 2048.0f);
    __syncthreads();
    red[threadIdx.x] = s2; __syncthreads();
    for (int o = 128; o; o >>= 1) {
        if (threadIdx.x < o) red[threadIdx.x] += red[threadIdx.x + o];
        __syncthreads();
    }
    float mean2 = red[0] * (1.0f / 2048.0f);
    float var = mean2 - mean * mean;
    float inv = rsqrtf(var + 1e-5f);
    for (int i = threadIdx.x; i < 2048; i += 256)
        xnb[(size_t)row * 2048 + i] = f2bf((xr[i] - mean) * inv * gamma[i]);
}

// ---------------------------------------------------------------------------
// 2) Tiled bf16 WMMA GEMM: C[M,N](f32) = A[M,K](bf16) @ B[K,N](bf16)
//    block tile 128x64, K-step 32, 8 waves, wave -> 16x64 strip (4 WMMA tiles)
//    A tile staged via global_load_async_to_lds_b128.
// ---------------------------------------------------------------------------
__global__ void gemm_bf16_kernel(const unsigned short* __restrict__ A,
                                 const unsigned short* __restrict__ B,
                                 float* __restrict__ C,
                                 int M, int N, int K) {
    __shared__ alignas(16) unsigned short As[128 * 32];   // [row][k]
    __shared__ alignas(16) unsigned short Bst[64 * 32];   // [col][k]  (K-major)
    const int t = threadIdx.x;
    const int wave = t >> 5, lane = t & 31;
    const int rowBase = blockIdx.y * 128;
    const int colBase = blockIdx.x * 64;
    const int wrow = wave * 16;

    v8f acc[4];
#pragma unroll
    for (int i = 0; i < 4; ++i)
#pragma unroll
        for (int j = 0; j < 8; ++j) acc[i][j] = 0.0f;

    // per-thread A-tile chunk coordinates (2 x uint4 per thread)
    const int ar0 = t >> 1;                 // rows 0..127 (two threads per row)
    const int as0 = (t & 1) * 2;            // uint4 segment 0/2

    for (int k0 = 0; k0 < K; k0 += 32) {
        __syncthreads();
        // A tile: 128x32 halves, async memory->LDS (2 b128 per thread)
#pragma unroll
        for (int i = 0; i < 2; ++i) {
            int sgi = as0 + i;
            async_copy_b128(&As[ar0 * 32 + sgi * 8],
                            A + (size_t)(rowBase + ar0) * K + k0 + sgi * 8);
        }
        // prefetch next A K-slice
        if (k0 + 32 < K)
            __builtin_prefetch(A + (size_t)(rowBase + ar0) * K + k0 + 32 + as0 * 8, 0, 0);
        // B tile transposed into K-major: 32x64 halves, 8 scalars per thread
#pragma unroll
        for (int i = 0; i < 8; ++i) {
            int l = t + i * 256;
            int kk = l >> 6, c = l & 63;
            Bst[c * 32 + kk] = B[(size_t)(k0 + kk) * N + colBase + c];
        }
        wait_async0();
        __syncthreads();

        v16bf afrag = load_a_frag(As + (wrow + (lane & 15)) * 32, lane);
        v16bf bfrag[4];
#pragma unroll
        for (int ct = 0; ct < 4; ++ct)
            bfrag[ct] = load_b_frag(Bst + (ct * 16 + (lane & 15)) * 32 + (lane >> 4) * 16);
#pragma unroll
        for (int ct = 0; ct < 4; ++ct)
            acc[ct] = wmma_bf16(afrag, bfrag[ct], acc[ct]);
    }

#pragma unroll
    for (int ct = 0; ct < 4; ++ct)
#pragma unroll
        for (int r = 0; r < 8; ++r) {
            int m = rowBase + wrow + ((lane >> 4) * 8) + r;
            int nn = colBase + ct * 16 + (lane & 15);
            C[(size_t)m * N + nn] = acc[ct][r];
        }
}

// ---------------------------------------------------------------------------
// 3) RoPE + scale; produce bf16 Q (b,h,n,dh), K (b,n,dh), V (b,n,dh)
// ---------------------------------------------------------------------------
__global__ void rotary_kernel(const float* __restrict__ qf,
                              const float* __restrict__ kvf,
                              unsigned short* __restrict__ qb,
                              unsigned short* __restrict__ kb,
                              unsigned short* __restrict__ vb) {
    const int tok = blockIdx.x;            // 0..4095
    const int b = tok >> 11, npos = tok & 2047;
    const float* qrow = qf + (size_t)tok * 2048;
    const float* kvrow = kvf + (size_t)tok * 256;
    const float scale = 0.08838834764831845f;        // 128^-0.5
    const float lnc = -0.14391156831212787f;         // -ln(10000)/64

    for (int idx = threadIdx.x; idx < 2048; idx += 256) {
        int h = idx >> 7, d = idx & 127;
        float invf = __expf((float)(d & 63) * lnc);
        float th = (float)npos * invf;
        float c = __cosf(th), sn = __sinf(th);
        float v = qrow[idx];
        float vp = qrow[(idx & ~127) | (d ^ 64)];
        float rot = (d < 64) ? -vp : vp;
        qb[(((size_t)(b * 16 + h)) * 2048 + npos) * 128 + d] = f2bf((v * c + rot * sn) * scale);
    }
    for (int d = threadIdx.x; d < 128; d += 256) {
        float invf = __expf((float)(d & 63) * lnc);
        float th = (float)npos * invf;
        float c = __cosf(th), sn = __sinf(th);
        float v = kvrow[d];
        float vp = kvrow[d ^ 64];
        float rot = (d < 64) ? -vp : vp;
        size_t o = ((size_t)b * 2048 + npos) * 128 + d;
        kb[o] = f2bf(v * c + rot * sn);
        vb[o] = f2bf(kvrow[128 + d]);
    }
}

// ---------------------------------------------------------------------------
// 4) Causal flash attention (multi-query): block = 128 q-rows x 1 head,
//    8 waves x 16 q-rows; key tiles of 32; online softmax; WMMA bf16.
//    Q and K tiles staged via global_load_async_to_lds_b128.
// ---------------------------------------------------------------------------
__global__ void attn_kernel(const unsigned short* __restrict__ qb,
                            const unsigned short* __restrict__ kb,
                            const unsigned short* __restrict__ vb,
                            unsigned short* __restrict__ attnb) {
    const int qtile = blockIdx.x, head = blockIdx.y, batch = blockIdx.z;
    __shared__ alignas(16) unsigned short Qs[128 * 128];  // [qrow][dh]
    __shared__ alignas(16) unsigned short Ks[32 * 128];   // [key][dh]   (K-major for S)
    __shared__ alignas(16) unsigned short Vt[128 * 32];   // [col][key]  (K-major for PV)
    __shared__ alignas(16) unsigned short Ps[8 * 16 * 32];// per-wave P scratch

    const int t = threadIdx.x, wave = t >> 5, lane = t & 31;
    const int qbase = qtile * 128;
    const unsigned short* qsrc = qb + (((size_t)(batch * 16 + head)) * 2048 + qbase) * 128;
    const unsigned short* ksrc = kb + ((size_t)batch * 2048) * 128;
    const unsigned short* vsrc = vb + ((size_t)batch * 2048) * 128;

    // stage Q block tile (128x128 bf16 = 2048 b128 chunks, 8 per thread, async)
#pragma unroll
    for (int i = 0; i < 8; ++i) {
        int l = t + i * 256;
        int r = l >> 4, sgi = l & 15;
        async_copy_b128(&Qs[r * 128 + sgi * 8], qsrc + (size_t)r * 128 + sgi * 8);
    }
    wait_async0();
    __syncthreads();

    v16bf qf[4];
    const unsigned short* qrow = Qs + (wave * 16 + (lane & 15)) * 128;
#pragma unroll
    for (int ks = 0; ks < 4; ++ks) qf[ks] = load_a_frag(qrow + ks * 32, lane);

    v8f acc[8];
#pragma unroll
    for (int i = 0; i < 8; ++i)
#pragma unroll
        for (int j = 0; j < 8; ++j) acc[i][j] = 0.0f;
    float mrow[8], lrow[8];
#pragma unroll
    for (int r = 0; r < 8; ++r) { mrow[r] = -1e30f; lrow[r] = 0.0f; }

    const int myM = (lane >> 4) * 8;
    const int nTiles = qtile * 4 + 4;   // causal: keys <= qbase+127

    for (int jt = 0; jt < nTiles; ++jt) {
        const int kb0 = jt * 32;
        __syncthreads();
        // K tile 32x128: async memory->LDS (2 b128 per thread)
#pragma unroll
        for (int i = 0; i < 2; ++i) {
            int l = t + i * 256;
            int r = l >> 4, sgi = l & 15;
            async_copy_b128(&Ks[r * 128 + sgi * 8],
                            ksrc + (size_t)(kb0 + r) * 128 + sgi * 8);
        }
        // V tile transposed (4096 halves, 16/thread)
#pragma unroll
        for (int i = 0; i < 16; ++i) {
            int l = t + i * 256;
            int key = l >> 7, c = l & 127;
            Vt[c * 32 + key] = vsrc[(size_t)(kb0 + key) * 128 + c];
        }
        wait_async0();
        __syncthreads();

        // S = Q @ K^T  (two 16x16 tiles per wave)
        v8f sA[2];
#pragma unroll
        for (int st = 0; st < 2; ++st) {
#pragma unroll
            for (int j = 0; j < 8; ++j) sA[st][j] = 0.0f;
            v16bf bfr[4];
#pragma unroll
            for (int ks = 0; ks < 4; ++ks)
                bfr[ks] = load_b_frag(Ks + (st * 16 + (lane & 15)) * 128 +
                                      ks * 32 + (lane >> 4) * 16);
#pragma unroll
            for (int ks = 0; ks < 4; ++ks)
                sA[st] = wmma_bf16(qf[ks], bfr[ks], sA[st]);
        }

        // causal mask + online softmax (rows striped per C-layout)
#pragma unroll
        for (int r = 0; r < 8; ++r) {
            int qg = qbase + wave * 16 + myM + r;
            float v0 = sA[0][r];
            float v1 = sA[1][r];
            if (kb0 + (lane & 15) > qg)       v0 = -1e30f;
            if (kb0 + 16 + (lane & 15) > qg)  v1 = -1e30f;
            float rm = fmaxf(v0, v1);
#pragma unroll
            for (int o = 8; o; o >>= 1) rm = fmaxf(rm, __shfl_xor(rm, o, 16));
            float mnew = fmaxf(mrow[r], rm);
            float corr = __expf(mrow[r] - mnew);
            float p0 = __expf(v0 - mnew);
            float p1 = __expf(v1 - mnew);
            float rs = p0 + p1;
#pragma unroll
            for (int o = 8; o; o >>= 1) rs += __shfl_xor(rs, o, 16);
            lrow[r] = lrow[r] * corr + rs;
            mrow[r] = mnew;
#pragma unroll
            for (int ot = 0; ot < 8; ++ot) acc[ot][r] *= corr;
            // spill P to per-wave LDS scratch in row-major (16x32)
            unsigned short* pp = Ps + wave * 512 + (myM + r) * 32;
            pp[(lane & 15)]      = f2bf(p0);
            pp[16 + (lane & 15)] = f2bf(p1);
        }

        // P (16x32) as WMMA A fragment, then acc += P @ V
        v16bf pf = load_a_frag(Ps + wave * 512 + (lane & 15) * 32, lane);
#pragma unroll
        for (int ot = 0; ot < 8; ++ot) {
            v16bf bf = load_b_frag(Vt + (ot * 16 + (lane & 15)) * 32 + (lane >> 4) * 16);
            acc[ot] = wmma_bf16(pf, bf, acc[ot]);
        }
    }

    // epilogue: normalize and store bf16 into (b, n, h*dh)
    const size_t orow0 = (size_t)batch * 2048 + qbase + wave * 16;
#pragma unroll
    for (int ot = 0; ot < 8; ++ot)
#pragma unroll
        for (int r = 0; r < 8; ++r) {
            size_t row = orow0 + myM + r;
            int col = head * 128 + ot * 16 + (lane & 15);
            attnb[row * 2048 + col] = f2bf(acc[ot][r] / lrow[r]);
        }
}

// ---------------------------------------------------------------------------
// launch
// ---------------------------------------------------------------------------
extern "C" void kernel_launch(void* const* d_in, const int* in_sizes, int n_in,
                              void* d_out, int out_size, void* d_ws, size_t ws_size,
                              hipStream_t stream) {
    const float* x     = (const float*)d_in[0];   // (2,2048,2048)
    const float* gamma = (const float*)d_in[1];   // (2048,)
    const float* Wq    = (const float*)d_in[2];   // (2048,2048)
    const float* Wkv   = (const float*)d_in[3];   // (2048,256)
    const float* Wo    = (const float*)d_in[4];   // (2048,2048)
    float* out = (float*)d_out;                   // (2,2048,2048)

    char* ws = (char*)d_ws;
    size_t off = 0;
    auto alloc = [&](size_t bytes) {
        void* p = ws + off;
        off += (bytes + 255) & ~(size_t)255;
        return p;
    };
    unsigned short* xnb  = (unsigned short*)alloc((size_t)4096 * 2048 * 2);
    unsigned short* wqb  = (unsigned short*)alloc((size_t)2048 * 2048 * 2);
    unsigned short* wkvb = (unsigned short*)alloc((size_t)2048 * 256 * 2);
    unsigned short* wob  = (unsigned short*)alloc((size_t)2048 * 2048 * 2);
    float*          qf   = (float*)alloc((size_t)4096 * 2048 * 4);
    float*          kvf  = (float*)alloc((size_t)4096 * 256 * 4);
    unsigned short* qbuf = (unsigned short*)alloc((size_t)2 * 16 * 2048 * 128 * 2);
    unsigned short* kbuf = (unsigned short*)alloc((size_t)2 * 2048 * 128 * 2);
    unsigned short* vbuf = (unsigned short*)alloc((size_t)2 * 2048 * 128 * 2);
    unsigned short* attb = (unsigned short*)alloc((size_t)4096 * 2048 * 2);

    cvt_bf16_kernel<<<2048, 256, 0, stream>>>(Wq,  wqb,  2048 * 2048);
    cvt_bf16_kernel<<<512,  256, 0, stream>>>(Wkv, wkvb, 2048 * 256);
    cvt_bf16_kernel<<<2048, 256, 0, stream>>>(Wo,  wob,  2048 * 2048);

    ln_kernel<<<4096, 256, 0, stream>>>(x, gamma, xnb);

    // Q = xn @ Wq : (4096 x 2048) = (4096 x 2048)(2048 x 2048)
    gemm_bf16_kernel<<<dim3(2048 / 64, 4096 / 128), 256, 0, stream>>>(
        xnb, wqb, qf, 4096, 2048, 2048);
    // KV = xn @ Wkv : (4096 x 256)
    gemm_bf16_kernel<<<dim3(256 / 64, 4096 / 128), 256, 0, stream>>>(
        xnb, wkvb, kvf, 4096, 256, 2048);

    rotary_kernel<<<4096, 256, 0, stream>>>(qf, kvf, qbuf, kbuf, vbuf);

    attn_kernel<<<dim3(16, 16, 2), 256, 0, stream>>>(qbuf, kbuf, vbuf, attb);

    // out = attn @ Wo : (4096 x 2048)
    gemm_bf16_kernel<<<dim3(2048 / 64, 4096 / 128), 256, 0, stream>>>(
        attb, wob, out, 4096, 2048, 2048);
}
